// GatedDeltaNet_19129784337097
// MI455X (gfx1250) — matrix operand order repaired and checked by reading, not compiled, not run on hardware
//
#include <hip/hip_runtime.h>
#include <hip/hip_bf16.h>

// GatedDeltaNet forward for MI455X (gfx1250, wave32, WMMA bf16 path).
// WMMA operands stored K-contiguous (A row-major, B transposed) -> b128
// fragment loads; batched B fragments; global_prefetch of next GEMM tiles;
// register-resident scan state; guarded hw bf16 conversion.

typedef __attribute__((ext_vector_type(16))) __bf16 v16bf;
typedef __attribute__((ext_vector_type(8)))  __bf16 v8bf;
typedef __attribute__((ext_vector_type(8)))  float  v8f;

#define DEV static __device__ __forceinline__

#if defined(__has_builtin)
#if __has_builtin(__builtin_amdgcn_cvt_pk_bf16_f32)
#define GDN_HAVE_CVT_PK_BF16 1
#endif
#endif

DEV unsigned short f2bf_sw(float f) {
  unsigned int u = __builtin_bit_cast(unsigned int, f);
  u += 0x7FFFu + ((u >> 16) & 1u);   // round-to-nearest-even
  return (unsigned short)(u >> 16);
}
DEV unsigned int pack2(float x, float y) {
#ifdef GDN_HAVE_CVT_PK_BF16
  auto r = __builtin_amdgcn_cvt_pk_bf16_f32(x, y);   // v_cvt_pk_bf16_f32
  return __builtin_bit_cast(unsigned int, r);
#else
  return (unsigned int)f2bf_sw(x) | ((unsigned int)f2bf_sw(y) << 16);
#endif
}
DEV unsigned short f2bf(float f) {
#ifdef GDN_HAVE_CVT_PK_BF16
  return (unsigned short)(pack2(f, 0.f) & 0xFFFFu);
#else
  return f2bf_sw(f);
#endif
}
DEV v16bf cat8(v8bf a, v8bf b) {
  return __builtin_shufflevector(a, b, 0, 1, 2, 3, 4, 5, 6, 7,
                                 8, 9, 10, 11, 12, 13, 14, 15);
}
// A operand (16x32): lane holds K {hi*8..+7} and {16+hi*8..+7} of its row.
DEV v16bf frag_a(const unsigned short* rowp, int hi) {
  v8bf a0 = *(const v8bf*)(rowp + hi * 8);
  v8bf a1 = *(const v8bf*)(rowp + 16 + hi * 8);
  return cat8(a0, a1);
}
// B operand (32x16) from K-transposed storage: lane holds K {hi*8..+7} and {16+hi*8..+7}.
DEV v16bf frag_b(const unsigned short* rowp, int hi) {
  v8bf b0 = *(const v8bf*)(rowp + hi * 8);
  v8bf b1 = *(const v8bf*)(rowp + 16 + hi * 8);
  return cat8(b0, b1);
}
DEV v8f wmma_bf(v16bf a, v16bf b, v8f c) {
  return __builtin_amdgcn_wmma_f32_16x16x32_bf16(false, a, false, b,
                                                 (short)0, c, false, false);
}

// ---------------------------------------------------------------- GEMM -----
// C = A(f32 MxK) * B(f32 KxN), fp32 accum via bf16 WMMA. 128x128 tile, K=32.
__global__ __launch_bounds__(256) void gdn_gemm_bf16(
    const float* __restrict__ A, const float* __restrict__ Bm,
    float* __restrict__ Cm, int M, int N, int K) {
  __shared__ alignas(16) unsigned short As[128][40];   // [m][k]
  __shared__ alignas(16) unsigned short Bt[128][40];   // [n][k] (transposed)
  const int tid = threadIdx.x;
  const int lane = tid & 31, wv = tid >> 5;
  const int hi = lane >> 4, l16 = lane & 15;
  const int m0 = blockIdx.y * 128, n0 = blockIdx.x * 128;
  v8f acc[8] = {};
  for (int k0 = 0; k0 < K; k0 += 32) {
    for (int idx = tid; idx < 1024; idx += 256) {       // A tile 128x32
      int r = idx >> 3, c4 = (idx & 7) * 4;
      float4 f = *(const float4*)(A + (size_t)(m0 + r) * K + k0 + c4);
      uint2 p; p.x = pack2(f.x, f.y); p.y = pack2(f.z, f.w);
      *(uint2*)&As[r][c4] = p;
    }
    for (int idx = tid; idx < 1024; idx += 256) {       // B tile 32x128 -> [n][k]
      int n = idx & 127, k4 = (idx >> 7) * 4;
      const float* bp = Bm + (size_t)(k0 + k4) * N + n0 + n;
      float b0 = bp[0], b1 = bp[N], b2 = bp[2 * (size_t)N], b3 = bp[3 * (size_t)N];
      uint2 p; p.x = pack2(b0, b1); p.y = pack2(b2, b3);
      *(uint2*)&Bt[n][k4] = p;
    }
    __syncthreads();
    if (k0 + 32 < K) {                                  // prefetch next tiles to L2
      if (tid < 128) {
        __builtin_prefetch(A + (size_t)(m0 + tid) * K + k0 + 32, 0, 1);
      } else {
        int t2 = tid - 128, kr = t2 >> 2, seg = (t2 & 3) * 32;
        __builtin_prefetch(Bm + (size_t)(k0 + 32 + kr) * N + n0 + seg, 0, 1);
      }
    }
    v16bf a = frag_a(&As[wv * 16 + l16][0], hi);
    v16bf bfr[8];
    #pragma unroll
    for (int nt = 0; nt < 8; ++nt) bfr[nt] = frag_b(&Bt[nt * 16 + l16][0], hi);
    #pragma unroll
    for (int nt = 0; nt < 8; ++nt) acc[nt] = wmma_bf(a, bfr[nt], acc[nt]);
    __syncthreads();
  }
  #pragma unroll
  for (int nt = 0; nt < 8; ++nt)
    #pragma unroll
    for (int r = 0; r < 8; ++r) {
      int row = m0 + wv * 16 + r + hi * 8;
      int col = n0 + nt * 16 + l16;
      Cm[(size_t)row * N + col] = acc[nt][r];
    }
}

// ------------------------------------------------------------- ba GEMM -----
__global__ void gdn_ba(const float* __restrict__ x, const float* __restrict__ Wba,
                       float* __restrict__ ba) {
  int idx = blockIdx.x * 256 + threadIdx.x;
  if (idx >= 8192 * 16) return;
  int j = idx & 15, bt = idx >> 4;
  const float* xr = x + (size_t)bt * 1024;
  float acc = 0.f;
  for (int c = 0; c < 1024; ++c) acc += xr[c] * Wba[c * 16 + j];
  ba[idx] = acc;
}

// ----------------------------------------------------- depthwise conv ------
__global__ __launch_bounds__(256) void gdn_conv(const float* __restrict__ qkvz,
                                                const float* __restrict__ w,
                                                float* __restrict__ out) {
  int idx = blockIdx.x * 256 + threadIdx.x;
  if (idx >= 2 * 4096 * 3072) return;
  int c = idx % 3072;
  int t = (idx / 3072) % 4096;
  int b = idx / (3072 * 4096);
  int seg = c >> 10, cc = c & 1023;
  int col = (cc >> 7) * 512 + seg * 128 + (cc & 127);   // map into qkvz width 4096
  float acc = 0.f;
  #pragma unroll
  for (int j = 0; j < 4; ++j) {
    int ts = t - 3 + j;
    if (ts >= 0)
      acc += w[c * 4 + j] * qkvz[((size_t)(b * 4096 + ts)) * 4096 + col];
  }
  out[idx] = acc / (1.f + __expf(-acc));                // SiLU
}

// ----------------------------------------------------------- gate ----------
__global__ void gdn_gate(const float* __restrict__ ba, const float* __restrict__ A_log,
                         const float* __restrict__ dt_bias, float* __restrict__ g,
                         float* __restrict__ beta) {
  int idx = blockIdx.x * 256 + threadIdx.x;
  if (idx >= 8192 * 8) return;
  int h = idx & 7;
  size_t bt = idx >> 3;
  float b_ = ba[bt * 16 + 2 * h];
  float a_ = ba[bt * 16 + 2 * h + 1];
  beta[idx] = 1.f / (1.f + __expf(-b_));
  float xs = a_ + dt_bias[h];
  float sp = (xs > 20.f) ? xs : log1pf(__expf(xs));
  g[idx] = -__expf(A_log[h]) * sp;
}

// -------------------------------------------------- per-chunk preprocess ---
__global__ __launch_bounds__(256) void gdn_chunkprep(
    const float* __restrict__ conv, const float* __restrict__ gG,
    const float* __restrict__ betaG, unsigned short* __restrict__ attnG,
    unsigned short* __restrict__ qgG, unsigned short* __restrict__ kdG,
    unsigned short* __restrict__ kcG, float* __restrict__ vnG,
    float* __restrict__ glG) {
  __shared__ alignas(16) float vs[64][128];             // v*beta -> v_new
  __shared__ alignas(16) float kgs[64][128];            // k_beta*exp(g) -> k_cum
  __shared__ alignas(16) unsigned short q_bf[64][136];  // A operands, [row][K]
  __shared__ alignas(16) unsigned short k_bf[64][136];  // B operand (K-transposed)
  __shared__ alignas(16) unsigned short kb_bf[64][136];
  __shared__ float tri[64][65];
  __shared__ float qpart[64][4], kpart[64][4];
  __shared__ float gc[64], bet[64], qn[64], kn[64];

  const int cb = blockIdx.x;               // (b*8+h)*64 + chunk
  const int b = cb >> 9, h = (cb >> 6) & 7, n = cb & 63;
  const int t0 = n * 64;
  const int tid = threadIdx.x;
  const int lane = tid & 31, wv = tid >> 5;
  const int hi = lane >> 4, l16 = lane & 15;

  // partial sums for row l2-norms (float4 loads, 4 threads per row)
  {
    int r = tid >> 2, d0 = (tid & 3) * 32;
    const float* base = conv + ((size_t)(b * 4096 + t0 + r)) * 3072 + h * 128 + d0;
    float sq = 0.f, sk = 0.f;
    for (int d = 0; d < 32; d += 4) {
      float4 qf = *(const float4*)(base + d);
      float4 kf = *(const float4*)(base + 1024 + d);
      sq += qf.x * qf.x + qf.y * qf.y + qf.z * qf.z + qf.w * qf.w;
      sk += kf.x * kf.x + kf.y * kf.y + kf.z * kf.z + kf.w * kf.w;
    }
    qpart[r][tid & 3] = sq;
    kpart[r][tid & 3] = sk;
  }
  if (tid < 64) {
    int t = t0 + tid;
    gc[tid] = gG[((size_t)(b * 4096 + t)) * 8 + h];
    bet[tid] = betaG[((size_t)(b * 4096 + t)) * 8 + h];
  }
  __syncthreads();
  if (tid < 64) {
    float sq = qpart[tid][0] + qpart[tid][1] + qpart[tid][2] + qpart[tid][3];
    float sk = kpart[tid][0] + kpart[tid][1] + kpart[tid][2] + kpart[tid][3];
    qn[tid] = rsqrtf(sq + 1e-6f) * 0.08838834764831845f;  // l2norm * dk^-0.5
    kn[tid] = rsqrtf(sk + 1e-6f);
  }
  __syncthreads();
  if (tid == 0) {                          // cumsum of g within chunk
    float s = 0.f;
    for (int i = 0; i < 64; ++i) { s += gc[i]; gc[i] = s; }
  }
  __syncthreads();
  const float glast = gc[63];

  for (int idx = tid; idx < 2048; idx += 256) {
    int r = idx >> 5, d4 = (idx & 31) << 2;
    size_t base = ((size_t)(b * 4096 + t0 + r)) * 3072 + h * 128 + d4;
    float4 qf = *(const float4*)(conv + base);
    float4 kf = *(const float4*)(conv + base + 1024);
    float4 vf = *(const float4*)(conv + base + 2048);
    float qnr = qn[r], knr = kn[r], br = bet[r];
    float eg = __expf(gc[r]);
    float ed = __expf(glast - gc[r]);
    float q0 = qf.x * qnr, q1 = qf.y * qnr, q2 = qf.z * qnr, q3 = qf.w * qnr;
    float k0 = kf.x * knr, k1 = kf.y * knr, k2 = kf.z * knr, k3 = kf.w * knr;
    uint2 p;
    p.x = pack2(q0, q1); p.y = pack2(q2, q3);
    *(uint2*)&q_bf[r][d4] = p;
    p.x = pack2(k0, k1); p.y = pack2(k2, k3);
    *(uint2*)&k_bf[r][d4] = p;
    p.x = pack2(k0 * br, k1 * br); p.y = pack2(k2 * br, k3 * br);
    *(uint2*)&kb_bf[r][d4] = p;
    float4 kg4 = {k0 * br * eg, k1 * br * eg, k2 * br * eg, k3 * br * eg};
    *(float4*)&kgs[r][d4] = kg4;
    float4 vb4 = {vf.x * br, vf.y * br, vf.z * br, vf.w * br};
    *(float4*)&vs[r][d4] = vb4;
    p.x = pack2(q0 * eg, q1 * eg); p.y = pack2(q2 * eg, q3 * eg);
    *(uint2*)(qgG + (size_t)cb * 8192 + r * 128 + d4) = p;
    p.x = pack2(k0 * ed, k1 * ed); p.y = pack2(k2 * ed, k3 * ed);
    *(uint2*)(kdG + (size_t)cb * 8192 + r * 128 + d4) = p;
  }
  if (tid == 0) glG[cb] = __expf(glast);
  __syncthreads();

  // tri = tril(k_beta @ k^T * decay, -1);  attn = tril(q @ k^T * decay)
  {
    const int mt = wv >> 1, nt0 = (wv & 1) * 2;
    const int arow = mt * 16 + l16;
    v8f at[2] = {}, aa[2] = {};
    for (int ks0 = 0; ks0 < 128; ks0 += 32) {
      v16bf akb = frag_a(&kb_bf[arow][ks0], hi);
      v16bf aq  = frag_a(&q_bf[arow][ks0], hi);
      v16bf bb0 = frag_b(&k_bf[(nt0 + 0) * 16 + l16][ks0], hi);
      v16bf bb1 = frag_b(&k_bf[(nt0 + 1) * 16 + l16][ks0], hi);
      at[0] = wmma_bf(akb, bb0, at[0]);
      aa[0] = wmma_bf(aq, bb0, aa[0]);
      at[1] = wmma_bf(akb, bb1, at[1]);
      aa[1] = wmma_bf(aq, bb1, aa[1]);
    }
    #pragma unroll
    for (int j = 0; j < 2; ++j)
      #pragma unroll
      for (int r = 0; r < 8; ++r) {
        int row = mt * 16 + r + hi * 8;
        int col = (nt0 + j) * 16 + l16;
        float dec = __expf(gc[row] - gc[col]);
        tri[row][col] = (col < row) ? at[j][r] * dec : 0.f;
        float av = (col <= row) ? aa[j][r] * dec : 0.f;
        attnG[(size_t)cb * 4096 + row * 64 + col] = f2bf(av);
      }
  }
  __syncthreads();

  // forward substitution: (I + tri)^{-1} @ [v_beta | k_beta*exp(g)]
  {
    float* colp = (tid < 128) ? &vs[0][tid] : &kgs[0][tid - 128];
    for (int i = 0; i < 64; ++i) {
      float x = colp[i * 128];
      for (int j = 0; j < i; ++j) x -= tri[i][j] * colp[j * 128];
      colp[i * 128] = x;
    }
  }
  __syncthreads();
  for (int idx = tid; idx < 2048; idx += 256) {
    int r = idx >> 5, d4 = (idx & 31) << 2;
    *(float4*)(vnG + (size_t)cb * 8192 + r * 128 + d4) = *(const float4*)&vs[r][d4];
    float4 kg = *(const float4*)&kgs[r][d4];
    uint2 p; p.x = pack2(kg.x, kg.y); p.y = pack2(kg.z, kg.w);
    *(uint2*)(kcG + (size_t)cb * 8192 + r * 128 + d4) = p;
  }
}

// ----------------------------------------------- sequential chunk scan -----
__global__ __launch_bounds__(256) void gdn_scan(
    const unsigned short* __restrict__ attnG, const unsigned short* __restrict__ qgG,
    const unsigned short* __restrict__ kdG, const unsigned short* __restrict__ kcG,
    const float* __restrict__ vnG, const float* __restrict__ glG,
    float* __restrict__ coreG) {
  __shared__ alignas(16) unsigned short SbfT[128][136]; // [v][k] = bf16(S[k][v])
  __shared__ alignas(16) unsigned short vnT[128][72];   // [v][c] = bf16(vn[c][v])
  __shared__ alignas(16) unsigned short kdT[128][72];   // [m][c] = bf16(kd[c][m])
  const int bh = blockIdx.x;                    // b*8 + h
  const int b = bh >> 3, h = bh & 7;
  const int tid = threadIdx.x;
  const int lane = tid & 31, wv = tid >> 5;
  const int hi = lane >> 4, l16 = lane & 15;
  const int mt = wv >> 1, ntb = (wv & 1) * 4;
  const int arow = mt * 16 + l16;

  for (int e = tid; e < 128 * 136; e += 256) (&SbfT[0][0])[e] = 0;
  v8f st[8] = {};   // wave-owned fp32 state strip: st[nt][r]=S[wv*16+r+hi*8][nt*16+l16]
  __syncthreads();

  for (int ch = 0; ch < 64; ++ch) {
    const size_t cb = (size_t)bh * 64 + ch;
    const unsigned short* attnC = attnG + cb * 4096;
    const unsigned short* qgC = qgG + cb * 8192;
    const unsigned short* kcC = kcG + cb * 8192;
    const float* vnC = vnG + cb * 8192;
    for (int e = tid; e < 8192; e += 256) {     // stage kd transposed
      int c = e >> 7, m = e & 127;
      kdT[m][c] = kdG[cb * 8192 + e];
    }
    __syncthreads();

    // (a) v_n = v_new - k_cum @ S   (store transposed bf16)
    {
      v8f acc[4] = {};
      const unsigned short* arp = kcC + arow * 128;
      for (int ks0 = 0; ks0 < 128; ks0 += 32) {
        v16bf a = frag_a(arp + ks0, hi);
        v16bf bb[4];
        #pragma unroll
        for (int j = 0; j < 4; ++j)
          bb[j] = frag_b(&SbfT[(ntb + j) * 16 + l16][ks0], hi);
        #pragma unroll
        for (int j = 0; j < 4; ++j) acc[j] = wmma_bf(a, bb[j], acc[j]);
      }
      #pragma unroll
      for (int j = 0; j < 4; ++j)
        #pragma unroll
        for (int r = 0; r < 8; ++r) {
          int row = mt * 16 + r + hi * 8;
          int col = (ntb + j) * 16 + l16;
          vnT[col][row] = f2bf(vnC[row * 128 + col] - acc[j][r]);
        }
    }
    __syncthreads();

    // (b) out = qg @ S + attn @ v_n
    {
      v8f oc[4] = {};
      const unsigned short* arp = qgC + arow * 128;
      for (int ks0 = 0; ks0 < 128; ks0 += 32) {
        v16bf a = frag_a(arp + ks0, hi);
        v16bf bb[4];
        #pragma unroll
        for (int j = 0; j < 4; ++j)
          bb[j] = frag_b(&SbfT[(ntb + j) * 16 + l16][ks0], hi);
        #pragma unroll
        for (int j = 0; j < 4; ++j) oc[j] = wmma_bf(a, bb[j], oc[j]);
      }
      const unsigned short* arp2 = attnC + arow * 64;
      for (int ks0 = 0; ks0 < 64; ks0 += 32) {
        v16bf a = frag_a(arp2 + ks0, hi);
        v16bf bb[4];
        #pragma unroll
        for (int j = 0; j < 4; ++j)
          bb[j] = frag_b(&vnT[(ntb + j) * 16 + l16][ks0], hi);
        #pragma unroll
        for (int j = 0; j < 4; ++j) oc[j] = wmma_bf(a, bb[j], oc[j]);
      }
      #pragma unroll
      for (int j = 0; j < 4; ++j)
        #pragma unroll
        for (int r = 0; r < 8; ++r) {
          int row = mt * 16 + r + hi * 8;
          int col = (ntb + j) * 16 + l16;
          coreG[(((size_t)(b * 4096 + ch * 64 + row)) * 8 + h) * 128 + col] = oc[j][r];
        }
    }
    __syncthreads();

    // (c) S = S * exp(g_last) + kd^T @ v_n   (state in registers, rank-64 WMMA)
    {
      const float gl = glG[cb];
      #pragma unroll
      for (int nt = 0; nt < 8; ++nt) st[nt] = st[nt] * gl;
      for (int ks0 = 0; ks0 < 64; ks0 += 32) {
        v16bf a = frag_a(&kdT[wv * 16 + l16][ks0], hi);
        v16bf bb[8];
        #pragma unroll
        for (int nt = 0; nt < 8; ++nt)
          bb[nt] = frag_b(&vnT[nt * 16 + l16][ks0], hi);
        #pragma unroll
        for (int nt = 0; nt < 8; ++nt) st[nt] = wmma_bf(a, bb[nt], st[nt]);
      }
      #pragma unroll
      for (int nt = 0; nt < 8; ++nt)
        #pragma unroll
        for (int r = 0; r < 8; ++r) {
          int row = wv * 16 + r + hi * 8;      // DK index
          int col = nt * 16 + l16;             // DV index
          SbfT[col][row] = f2bf(st[nt][r]);
        }
    }
    __syncthreads();
  }
}

// ------------------------------------------------- gated RMSNorm -----------
__global__ __launch_bounds__(128) void gdn_norm(const float* __restrict__ core,
                                                const float* __restrict__ qkvz,
                                                const float* __restrict__ nw,
                                                float* __restrict__ hg) {
  int row = blockIdx.x;                 // 0 .. B*T*H-1
  int d = threadIdx.x;                  // 0..127
  int h = row & 7;
  size_t bt = (size_t)(row >> 3);
  float v = core[(size_t)row * 128 + d];
  __shared__ float red[128];
  red[d] = v * v;
  __syncthreads();
  for (int s = 64; s > 0; s >>= 1) {
    if (d < s) red[d] += red[d + s];
    __syncthreads();
  }
  float var = red[0] * (1.f / 128.f);
  float z = qkvz[bt * 4096 + h * 512 + 384 + d];
  float sz = z / (1.f + __expf(-z));
  hg[bt * 1024 + h * 128 + d] = nw[d] * v * rsqrtf(var + 1e-6f) * sz;
}

// ---------------------------------------------------------------------------
extern "C" void kernel_launch(void* const* d_in, const int* in_sizes, int n_in,
                              void* d_out, int out_size, void* d_ws, size_t ws_size,
                              hipStream_t stream) {
  (void)in_sizes; (void)n_in; (void)out_size; (void)ws_size;
  const float* x       = (const float*)d_in[0];
  const float* W_qkvz  = (const float*)d_in[1];
  const float* W_ba    = (const float*)d_in[2];
  const float* conv_w  = (const float*)d_in[3];
  const float* dt_bias = (const float*)d_in[4];
  const float* A_log   = (const float*)d_in[5];
  const float* norm_w  = (const float*)d_in[6];
  const float* W_out   = (const float*)d_in[7];
  float* out = (float*)d_out;

  char* ws = (char*)d_ws;
  size_t o = 0;
  auto carve = [&](size_t bytes) { size_t cur = o; o += (bytes + 255) & ~(size_t)255; return cur; };
  float* qkvz = (float*)(ws + carve((size_t)8192 * 4096 * 4));
  float* ba   = (float*)(ws + carve((size_t)8192 * 16 * 4));
  float* conv = (float*)(ws + carve((size_t)8192 * 3072 * 4));
  float* g    = (float*)(ws + carve((size_t)8192 * 8 * 4));
  float* beta = (float*)(ws + carve((size_t)8192 * 8 * 4));
  unsigned short* attn = (unsigned short*)(ws + carve((size_t)1024 * 4096 * 2));
  unsigned short* qg   = (unsigned short*)(ws + carve((size_t)1024 * 8192 * 2));
  unsigned short* kd   = (unsigned short*)(ws + carve((size_t)1024 * 8192 * 2));
  unsigned short* kc   = (unsigned short*)(ws + carve((size_t)1024 * 8192 * 2));
  float* vnew = (float*)(ws + carve((size_t)1024 * 8192 * 4));
  float* gl   = (float*)(ws + carve((size_t)1024 * 4));
  float* core = (float*)(ws + carve((size_t)8192 * 1024 * 4));
  float* hg   = (float*)(ws + carve((size_t)8192 * 1024 * 4));

  gdn_gemm_bf16<<<dim3(4096 / 128, 8192 / 128), 256, 0, stream>>>(
      x, W_qkvz, qkvz, 8192, 4096, 1024);
  gdn_ba<<<(8192 * 16 + 255) / 256, 256, 0, stream>>>(x, W_ba, ba);
  gdn_conv<<<(2 * 4096 * 3072 + 255) / 256, 256, 0, stream>>>(qkvz, conv_w, conv);
  gdn_gate<<<(8192 * 8 + 255) / 256, 256, 0, stream>>>(ba, A_log, dt_bias, g, beta);
  gdn_chunkprep<<<1024, 256, 0, stream>>>(conv, g, beta, attn, qg, kd, kc, vnew, gl);
  gdn_scan<<<16, 256, 0, stream>>>(attn, qg, kd, kc, vnew, gl, core);
  gdn_norm<<<8192 * 8, 128, 0, stream>>>(core, qkvz, norm_w, hg);
  gdn_gemm_bf16<<<dim3(1024 / 128, 8192 / 128), 256, 0, stream>>>(
      hg, W_out, out, 8192, 1024, 1024);
}
